// MultiAnchorFusionModel_37237366456824
// MI455X (gfx1250) — matrix operand order repaired and checked by reading, not compiled, not run on hardware
//
#include <hip/hip_runtime.h>
#include <hip/hip_bf16.h>
#include <math.h>

// ---------------------------------------------------------------------------
// Types for CDNA5 WMMA / TDM
// ---------------------------------------------------------------------------
typedef __bf16 bf16_t;
typedef __attribute__((ext_vector_type(16))) __bf16 v16bf;
typedef __attribute__((ext_vector_type(8)))  float  v8f;
typedef __attribute__((ext_vector_type(4)))  unsigned int v4u;
typedef __attribute__((ext_vector_type(8)))  int    v8i;
typedef __attribute__((ext_vector_type(4)))  int    v4i;

#if defined(__HIP_DEVICE_COMPILE__) && __has_builtin(__builtin_amdgcn_tensor_load_to_lds)
#define HAVE_TDM 1
#else
#define HAVE_TDM 0
#endif

#define DEVFN static __device__ __forceinline__

DEVFN v8f wmma_bf16(v16bf a, v16bf b, v8f c) {
  // D = A(16x32 bf16) * B(32x16 bf16) + C(16x16 f32)
  return __builtin_amdgcn_wmma_f32_16x16x32_bf16(
      /*neg_a=*/false, a, /*neg_b=*/false, b,
      /*c_mod=*/(short)0, c, /*reuse_a=*/false, /*reuse_b=*/false);
}

// A fragment: 16x32 bf16, element (m,k) = src[m*ld + k]
DEVFN v16bf load_a_bf16(const bf16_t* p, int ld, int lane) {
  const int m = lane & 15, half = lane >> 4;
  const bf16_t* row = p + (size_t)m * ld;
  v16bf a;
#pragma unroll
  for (int v = 0; v < 8; ++v) {
    int k = ((v < 4) ? 2 * v : 16 + 2 * (v - 4)) + half * 8;
    a[2 * v]     = row[k];
    a[2 * v + 1] = row[k + 1];
  }
  return a;
}

// Same but f32 source, converted to bf16 on load.
DEVFN v16bf load_a_f32(const float* p, int ld, int lane) {
  const int m = lane & 15, half = lane >> 4;
  const float* row = p + (size_t)m * ld;
  v16bf a;
#pragma unroll
  for (int v = 0; v < 8; ++v) {
    int k = ((v < 4) ? 2 * v : 16 + 2 * (v - 4)) + half * 8;
    a[2 * v]     = (bf16_t)row[k];
    a[2 * v + 1] = (bf16_t)row[k + 1];
  }
  return a;
}

// B fragment from n-major source: element (k,n) = src[n*ld + k]
DEVFN v16bf load_bt_bf16(const bf16_t* p, int ld, int lane) {
  const int n = lane & 15, half = lane >> 4;
  const bf16_t* row = p + (size_t)n * ld + half * 8;
  v16bf b;
#pragma unroll
  for (int v = 0; v < 8; ++v) {
    int k = (v < 4) ? 2 * v : 16 + 2 * (v - 4);
    b[2 * v]     = row[k];
    b[2 * v + 1] = row[k + 1];
  }
  return b;
}

#if HAVE_TDM
// ---------------------------------------------------------------------------
// TDM: async 2D tile load (2-byte elements) global -> LDS.
// tile: tile1 rows x tile0 elements, row stride = stride0 elements.
// Lands row-major [tile1][tile0] at ldsOff. Tracked by TENSORcnt.
// D# layout per CDNA5 ISA.
// 6-arg builtin form (clang-23 / therock-10.0 lane).
// ---------------------------------------------------------------------------
DEVFN void tdm_load_2d_b16(const void* gaddr, unsigned ldsOff,
                           int tile0, int tile1, long stride0) {
  unsigned long long ga = (unsigned long long)gaddr;
  const unsigned td0 = 0x100000u, td1 = 0x100000u;  // huge tensor dims: no OOB clip
  v4u g0;
  g0[0] = 1u;                                        // count=1 (valid user D#)
  g0[1] = ldsOff;                                    // LDS byte address
  g0[2] = (unsigned)ga;                              // global addr [31:0]
  g0[3] = (unsigned)((ga >> 32) & 0x1FFFFFFu) | (2u << 30);  // addr[56:32] | type=2
  v8i g1;
  g1[0] = (int)(1u << 16);                           // wg_mask=0, data_size=1 (2B)
  g1[1] = (int)((td0 & 0xFFFFu) << 16);              // barrier_addr=0 | tensor_dim0 lo
  g1[2] = (int)(((td0 >> 16) & 0xFFFFu) | ((td1 & 0xFFFFu) << 16));
  g1[3] = (int)(((td1 >> 16) & 0xFFFFu) | (((unsigned)tile0 & 0xFFFFu) << 16));
  g1[4] = (int)((unsigned)tile1 & 0xFFFFu);          // tile_dim1 | tile_dim2=0
  g1[5] = (int)((unsigned long long)stride0 & 0xFFFFFFFFu);   // dim0_stride lo32
  g1[6] = (int)(((unsigned long long)stride0 >> 32) & 0xFFFFu); // hi16 | dim1_stride lo=0
  g1[7] = 0;                                         // dim1_stride hi
  v4i g2 = {0, 0, 0, 0};
  v4i g3 = {0, 0, 0, 0};
  v8i g4 = {0, 0, 0, 0, 0, 0, 0, 0};
  __builtin_amdgcn_tensor_load_to_lds(g0, g1, g2, g3, g4, /*cpol=*/0);
}
#endif

// ---------------------------------------------------------------------------
// f32 -> bf16 conversion (weights)
// ---------------------------------------------------------------------------
__global__ void f2b_kernel(const float* __restrict__ in, bf16_t* __restrict__ out, int n) {
  int i = blockIdx.x * blockDim.x + threadIdx.x;
  if (i < n) out[i] = (bf16_t)in[i];
}

// ---------------------------------------------------------------------------
// Pool by 4 from f32 [nMat][128][256][256] -> bf16 [nMat][64*64][128]
// ---------------------------------------------------------------------------
__global__ void pool4_kernel(const float* __restrict__ in, bf16_t* __restrict__ out, int nMat) {
  int id = blockIdx.x * blockDim.x + threadIdx.x;
  int total = nMat * 4096 * 128;
  if (id >= total) return;
  int c = id & 127;
  int p = (id >> 7) & 4095;
  int mat = id >> 19;
  int y = p >> 6, x = p & 63;
  const float* b = in + ((size_t)mat * 128 + c) * 65536 + (size_t)(y * 4) * 256 + x * 4;
  __builtin_prefetch(b + 4 * 256, 0, 0);
  float s = 0.f;
#pragma unroll
  for (int dy = 0; dy < 4; ++dy)
#pragma unroll
    for (int dx = 0; dx < 4; ++dx) s += b[dy * 256 + dx];
  out[id] = (bf16_t)(s * 0.0625f);
}

// ---------------------------------------------------------------------------
// Pool by 2 from bf16 [nMat][Wi*Wi][128] -> bf16 [nMat][Wo*Wo][128], Wi=2*Wo
// ---------------------------------------------------------------------------
__global__ void pool2_kernel(const bf16_t* __restrict__ in, bf16_t* __restrict__ out,
                             int nMat, int Wo) {
  int id = blockIdx.x * blockDim.x + threadIdx.x;
  int total = nMat * Wo * Wo * 128;
  if (id >= total) return;
  int c = id & 127;
  int p = (id >> 7) % (Wo * Wo);
  int mat = (id >> 7) / (Wo * Wo);
  int y = p / Wo, x = p % Wo;
  int Wi = Wo * 2;
  const bf16_t* b = in + ((size_t)mat * Wi * Wi + (size_t)(2 * y) * Wi + 2 * x) * 128 + c;
  float s = (float)b[0] + (float)b[128] +
            (float)b[(size_t)Wi * 128] + (float)b[(size_t)Wi * 128 + 128];
  out[((size_t)mat * Wo * Wo + p) * 128 + c] = (bf16_t)(s * 0.25f);
}

// ---------------------------------------------------------------------------
// Strip WMMA GEMM:  out[M,128] = X[M,K] * W[128,K]^T + bias
// One wave computes a 16x128 strip (8 accumulators, A reused across 8 WMMAs).
// Each 32x128 W k-panel is staged into LDS once per block (TDM async tensor
// load issued by wave 0, S_WAIT_TENSORCNT, then block barrier) and consumed
// by all 8 waves -> 8x less W traffic.
//   a_f32: A operand is f32 (converted on load)
//   mode 0: bf16 out[m*ldo + n]     (row-major activations)
//   mode 1: bf16 out[n*ldo + m]     (transposed, for V projection)
//   mode 2: f32  out[n*ldo + m]     (transposed, final [C][HW] output)
//   mode 3: f32  out[m*ldo + n] +=  (fusion accumulation)
//   mode 4: f32  out[m*ldo + n] =   (fusion first scale, with bias)
// Launch invariant: (M/16) % 8 == 0 (always true here).
// ---------------------------------------------------------------------------
__global__ void gemm_wmma_kernel(const void* __restrict__ Xp, int a_f32, int ldx,
                                 const bf16_t* __restrict__ W, int ldw,
                                 const float* __restrict__ bias,
                                 void* __restrict__ out, int ldo,
                                 int M, int K, int mode) {
  __shared__ bf16_t wbuf[128 * 32];  // W k-panel, [n=128][k=32] row-major
  int tid = threadIdx.x;
  int lane = tid & 31;
  int wv = tid >> 5;
  int wave = blockIdx.x * 8 + wv;
  bool active = wave < (M >> 4);
  int l0 = active ? (wave << 4) : 0;

  v8f acc[8];
#pragma unroll
  for (int t = 0; t < 8; ++t)
#pragma unroll
    for (int r = 0; r < 8; ++r) acc[t][r] = 0.f;

  for (int k0 = 0; k0 < K; k0 += 32) {
#if HAVE_TDM
    if (wv == 0) {
      unsigned ldsOff = (unsigned)(unsigned long long)(void*)&wbuf[0];  // flat->LDS offset
      tdm_load_2d_b16(W + k0, ldsOff, /*tile0=*/32, /*tile1=*/128, /*stride0=*/ldw);
      __builtin_amdgcn_s_wait_tensorcnt(0);
    }
#else
    {  // cooperative fallback: 256 threads x 16 contiguous bf16 (32B)
      int n = tid >> 1, kk = (tid & 1) * 16;
#pragma unroll
      for (int j = 0; j < 16; ++j)
        wbuf[n * 32 + kk + j] = W[(size_t)n * ldw + k0 + kk + j];
    }
#endif
    __syncthreads();

    v16bf a = a_f32
        ? load_a_f32((const float*)Xp + (size_t)l0 * ldx + k0, ldx, lane)
        : load_a_bf16((const bf16_t*)Xp + (size_t)l0 * ldx + k0, ldx, lane);
#pragma unroll
    for (int t = 0; t < 8; ++t) {
      v16bf b = load_bt_bf16(wbuf + t * 16 * 32, 32, lane);
      acc[t] = wmma_bf16(a, b, acc[t]);
    }
    __syncthreads();  // protect wbuf before next panel
  }

  if (!active) return;
  int n = lane & 15, half = lane >> 4;
#pragma unroll
  for (int t = 0; t < 8; ++t) {
    int c0 = t * 16;
    float bv_ = bias ? bias[c0 + n] : 0.f;
#pragma unroll
    for (int r = 0; r < 8; ++r) {
      int m = l0 + r + half * 8;
      float v = acc[t][r] + bv_;
      if (mode == 0)      ((bf16_t*)out)[(size_t)m * ldo + c0 + n] = (bf16_t)v;
      else if (mode == 1) ((bf16_t*)out)[(size_t)(c0 + n) * ldo + m] = (bf16_t)v;
      else if (mode == 2) ((float*)out)[(size_t)(c0 + n) * ldo + m] = v;
      else if (mode == 3) ((float*)out)[(size_t)m * ldo + c0 + n] += v;
      else                ((float*)out)[(size_t)m * ldo + c0 + n] = v;
    }
  }
}

// ---------------------------------------------------------------------------
// Flash-style multi-anchor attention for one scale, LDS-staged K/V chunks.
//   Q : bf16 [L][128]       (channel = head*32 + d)
//   K : bf16 [nA][L][128]   (row-major)
//   Vt: bf16 [nA][128][L]   (transposed: d-major)
//   out: f32 [L][128] = mean_n softmax(QK^T/sqrt(32)) V
// Block: 256 threads = 8 waves on one head; waves share each 32-m chunk.
// Requires L % 128 == 0 (true for L in {256,1024,4096}).
// ---------------------------------------------------------------------------
__global__ void attn_wmma_kernel(const bf16_t* __restrict__ Q,
                                 const bf16_t* __restrict__ K,
                                 const bf16_t* __restrict__ Vt,
                                 float* __restrict__ outp, int L, int nA) {
  __shared__ bf16_t kbuf[32 * 32];      // [m_local][d]
  __shared__ bf16_t vbuf[32 * 32];      // [d][m_local]
  __shared__ bf16_t pbuf[8 * 16 * 32];  // per-wave P staging (C-layout -> A-layout)

  int tid = threadIdx.x;
  int lane = tid & 31;
  int wv = tid >> 5;
  int head = blockIdx.y;
  int l0 = (blockIdx.x * 8 + wv) * 16;
  int nidx = lane & 15, half = lane >> 4;
  bf16_t* pb = pbuf + wv * 512;
  const float sc = 0.17677669529663687f;  // 1/sqrt(32)

  int srow = tid >> 3;          // 0..31
  int scol = (tid & 7) * 4;     // 0,4,...,28

  v16bf aq = load_a_bf16(Q + (size_t)l0 * 128 + head * 32, 128, lane);

  float os0[8], os1[8];
#pragma unroll
  for (int r = 0; r < 8; ++r) { os0[r] = 0.f; os1[r] = 0.f; }

  for (int an = 0; an < nA; ++an) {
    const bf16_t* Kn  = K  + (size_t)an * L * 128;
    const bf16_t* Vtn = Vt + (size_t)an * 128 * L;
    float mrun[8], lrun[8];
    v8f o0, o1;
#pragma unroll
    for (int r = 0; r < 8; ++r) { mrun[r] = -1e30f; lrun[r] = 0.f; o0[r] = 0.f; o1[r] = 0.f; }

    for (int m0 = 0; m0 < L; m0 += 32) {
      *(uint2*)&kbuf[srow * 32 + scol] =
          *(const uint2*)&Kn[(size_t)(m0 + srow) * 128 + head * 32 + scol];
      *(uint2*)&vbuf[srow * 32 + scol] =
          *(const uint2*)&Vtn[(size_t)(head * 32 + srow) * L + m0 + scol];
      __syncthreads();

      v8f z0, z1;
#pragma unroll
      for (int r = 0; r < 8; ++r) { z0[r] = 0.f; z1[r] = 0.f; }
      v8f s0 = wmma_bf16(aq, load_bt_bf16(kbuf, 32, lane), z0);
      v8f s1 = wmma_bf16(aq, load_bt_bf16(kbuf + 16 * 32, 32, lane), z1);

      float cand[8];
#pragma unroll
      for (int r = 0; r < 8; ++r) {
        s0[r] *= sc; s1[r] *= sc;
        cand[r] = fmaxf(s0[r], s1[r]);
      }
#pragma unroll
      for (int off = 1; off < 16; off <<= 1)
#pragma unroll
        for (int r = 0; r < 8; ++r)
          cand[r] = fmaxf(cand[r], __shfl_xor(cand[r], off));

      float corr[8], rs[8];
#pragma unroll
      for (int r = 0; r < 8; ++r) {
        float mn = fmaxf(mrun[r], cand[r]);
        corr[r] = __expf(mrun[r] - mn);
        mrun[r] = mn;
        float p0 = __expf(s0[r] - mn);
        float p1 = __expf(s1[r] - mn);
        pb[(r + half * 8) * 32 + nidx]      = (bf16_t)p0;
        pb[(r + half * 8) * 32 + 16 + nidx] = (bf16_t)p1;
        rs[r] = p0 + p1;
      }
#pragma unroll
      for (int off = 1; off < 16; off <<= 1)
#pragma unroll
        for (int r = 0; r < 8; ++r)
          rs[r] += __shfl_xor(rs[r], off);
#pragma unroll
      for (int r = 0; r < 8; ++r) {
        lrun[r] = lrun[r] * corr[r] + rs[r];
        o0[r] *= corr[r]; o1[r] *= corr[r];
      }

      v16bf pa = load_a_bf16(pb, 32, lane);
      o0 = wmma_bf16(pa, load_bt_bf16(vbuf, 32, lane), o0);
      o1 = wmma_bf16(pa, load_bt_bf16(vbuf + 16 * 32, 32, lane), o1);

      __syncthreads();
    }
#pragma unroll
    for (int r = 0; r < 8; ++r) {
      float inv = 1.f / lrun[r];
      os0[r] += o0[r] * inv;
      os1[r] += o1[r] * inv;
    }
  }

  float invA = 1.f / (float)nA;
#pragma unroll
  for (int r = 0; r < 8; ++r) {
    int m = l0 + r + half * 8;
    outp[(size_t)m * 128 + head * 32 + nidx]      = os0[r] * invA;
    outp[(size_t)m * 128 + head * 32 + 16 + nidx] = os1[r] * invA;
  }
}

// ---------------------------------------------------------------------------
// Bilinear upsample (align_corners=False) f32 [Ws*Ws][128] -> bf16 [65536][128]
// ---------------------------------------------------------------------------
__global__ void upsample_kernel(const float* __restrict__ src, bf16_t* __restrict__ dst, int Ws) {
  int id = blockIdx.x * blockDim.x + threadIdx.x;
  if (id >= 65536 * 128) return;
  int c = id & 127;
  int p = id >> 7;
  int y = p >> 8, x = p & 255;
  float r = (float)Ws * (1.0f / 256.0f);
  float fy = (y + 0.5f) * r - 0.5f;
  float fx = (x + 0.5f) * r - 0.5f;
  int y0 = (int)floorf(fy);
  int x0 = (int)floorf(fx);
  float wy = fy - (float)y0, wx = fx - (float)x0;
  int y0c = y0 < 0 ? 0 : y0;
  int y1c = (y0 + 1 > Ws - 1) ? Ws - 1 : y0 + 1;
  int x0c = x0 < 0 ? 0 : x0;
  int x1c = (x0 + 1 > Ws - 1) ? Ws - 1 : x0 + 1;
  float v00 = src[((size_t)(y0c * Ws + x0c)) * 128 + c];
  float v01 = src[((size_t)(y0c * Ws + x1c)) * 128 + c];
  float v10 = src[((size_t)(y1c * Ws + x0c)) * 128 + c];
  float v11 = src[((size_t)(y1c * Ws + x1c)) * 128 + c];
  float v = (1.f - wy) * ((1.f - wx) * v00 + wx * v01) +
            wy * ((1.f - wx) * v10 + wx * v11);
  dst[id] = (bf16_t)v;
}

// ---------------------------------------------------------------------------
// Host launch
// ---------------------------------------------------------------------------
extern "C" void kernel_launch(void* const* d_in, const int* in_sizes, int n_in,
                              void* d_out, int out_size, void* d_ws, size_t ws_size,
                              hipStream_t stream) {
  (void)in_sizes; (void)n_in; (void)out_size;
  const float* query = (const float*)d_in[0];
  const float* keys  = (const float*)d_in[1];
  const float* vals  = (const float*)d_in[2];
  const float* wq = (const float*)d_in[3];
  const float* bq = (const float*)d_in[4];
  const float* wk = (const float*)d_in[5];
  const float* bk = (const float*)d_in[6];
  const float* wv = (const float*)d_in[7];
  const float* bv = (const float*)d_in[8];
  const float* fw = (const float*)d_in[9];
  const float* fb = (const float*)d_in[10];
  const float* ow = (const float*)d_in[11];
  const float* ob = (const float*)d_in[12];
  float* out = (float*)d_out;

  const int Ls[3]   = {256, 1024, 4096};   // si: 0->s16, 1->s8, 2->s4
  const int Wss[3]  = {16, 32, 64};
  const int offL[3] = {0, 256, 1280};
  const size_t kvoff[3] = {0, (size_t)3 * 128 * 256, (size_t)3 * 128 * (256 + 1024)};

  char* ws = (char*)d_ws;
  size_t pos = 0;
  auto alloc = [&](size_t bytes) -> void* {
    void* p = ws + pos;
    pos += (bytes + 255) & ~(size_t)255;
    return p;
  };
  bf16_t* wq_b = (bf16_t*)alloc((size_t)3 * 16384 * 2);
  bf16_t* wk_b = (bf16_t*)alloc((size_t)3 * 16384 * 2);
  bf16_t* wv_b = (bf16_t*)alloc((size_t)3 * 16384 * 2);
  bf16_t* fw_b = (bf16_t*)alloc((size_t)128 * 384 * 2);
  bf16_t* ow_b = (bf16_t*)alloc((size_t)128 * 128 * 2);
  bf16_t* Xq  = (bf16_t*)alloc((size_t)5376 * 128 * 2);
  bf16_t* Xk  = (bf16_t*)alloc((size_t)3 * 5376 * 128 * 2);
  bf16_t* Xv  = (bf16_t*)alloc((size_t)3 * 5376 * 128 * 2);
  bf16_t* Qp  = (bf16_t*)alloc((size_t)5376 * 128 * 2);
  bf16_t* Kp  = (bf16_t*)alloc((size_t)3 * 5376 * 128 * 2);   // row-major [L][128]
  bf16_t* Vtp = (bf16_t*)alloc((size_t)3 * 5376 * 128 * 2);   // transposed [128][L]
  bf16_t* Up  = (bf16_t*)alloc((size_t)65536 * 128 * 2);
  float*  attnb = (float*)alloc((size_t)5376 * 128 * 4);
  float*  Y1  = (float*)alloc((size_t)65536 * 128 * 4);
  if (pos > ws_size) return;

  auto cdiv = [](int a, int b) { return (a + b - 1) / b; };
  const int T = 256;

  // 1) weights -> bf16
  f2b_kernel<<<cdiv(3 * 16384, T), T, 0, stream>>>(wq, wq_b, 3 * 16384);
  f2b_kernel<<<cdiv(3 * 16384, T), T, 0, stream>>>(wk, wk_b, 3 * 16384);
  f2b_kernel<<<cdiv(3 * 16384, T), T, 0, stream>>>(wv, wv_b, 3 * 16384);
  f2b_kernel<<<cdiv(128 * 384, T), T, 0, stream>>>(fw, fw_b, 128 * 384);
  f2b_kernel<<<cdiv(128 * 128, T), T, 0, stream>>>(ow, ow_b, 128 * 128);

  // 2) pooling: factor 4 from inputs, then 2x hierarchical
  pool4_kernel<<<cdiv(1 * 4096 * 128, T), T, 0, stream>>>(query, Xq + (size_t)offL[2] * 128, 1);
  pool4_kernel<<<cdiv(3 * 4096 * 128, T), T, 0, stream>>>(keys,  Xk + kvoff[2], 3);
  pool4_kernel<<<cdiv(3 * 4096 * 128, T), T, 0, stream>>>(vals,  Xv + kvoff[2], 3);
  pool2_kernel<<<cdiv(1 * 1024 * 128, T), T, 0, stream>>>(Xq + (size_t)offL[2] * 128, Xq + (size_t)offL[1] * 128, 1, 32);
  pool2_kernel<<<cdiv(1 * 256 * 128, T), T, 0, stream>>>(Xq + (size_t)offL[1] * 128, Xq + (size_t)offL[0] * 128, 1, 16);
  pool2_kernel<<<cdiv(3 * 1024 * 128, T), T, 0, stream>>>(Xk + kvoff[2], Xk + kvoff[1], 3, 32);
  pool2_kernel<<<cdiv(3 * 256 * 128, T), T, 0, stream>>>(Xk + kvoff[1], Xk + kvoff[0], 3, 16);
  pool2_kernel<<<cdiv(3 * 1024 * 128, T), T, 0, stream>>>(Xv + kvoff[2], Xv + kvoff[1], 3, 32);
  pool2_kernel<<<cdiv(3 * 256 * 128, T), T, 0, stream>>>(Xv + kvoff[1], Xv + kvoff[0], 3, 16);

  // 3) QKV projections (strip WMMA GEMMs with TDM-staged W panels)
  for (int si = 0; si < 3; ++si) {
    int L = Ls[si];
    int blocks = cdiv(L / 16, 8);
    gemm_wmma_kernel<<<blocks, T, 0, stream>>>(
        Xq + (size_t)offL[si] * 128, 0, 128, wq_b + (size_t)si * 16384, 128,
        bq + si * 128, Qp + (size_t)offL[si] * 128, 128, L, 128, 0);
    for (int an = 0; an < 3; ++an) {
      gemm_wmma_kernel<<<blocks, T, 0, stream>>>(
          Xk + kvoff[si] + (size_t)an * L * 128, 0, 128, wk_b + (size_t)si * 16384, 128,
          bk + si * 128, Kp + kvoff[si] + (size_t)an * L * 128, 128, L, 128, 0);
      gemm_wmma_kernel<<<blocks, T, 0, stream>>>(
          Xv + kvoff[si] + (size_t)an * L * 128, 0, 128, wv_b + (size_t)si * 16384, 128,
          bv + si * 128, Vtp + kvoff[si] + (size_t)an * 128 * L, L, L, 128, 1);
    }
  }

  // 4) attention per scale (LDS-staged flash-softmax WMMA, mean over anchors)
  for (int si = 0; si < 3; ++si) {
    int L = Ls[si];
    dim3 grid(L / 128, 4);
    attn_wmma_kernel<<<grid, T, 0, stream>>>(
        Qp + (size_t)offL[si] * 128, Kp + kvoff[si], Vtp + kvoff[si],
        attnb + (size_t)offL[si] * 128, L, 3);
  }

  // 5) upsample each scale + fusion conv accumulation (WMMA)
  const int fblocks = cdiv(65536 / 16, 8);
  for (int si = 0; si < 3; ++si) {
    upsample_kernel<<<cdiv(65536 * 128, T), T, 0, stream>>>(
        attnb + (size_t)offL[si] * 128, Up, Wss[si]);
    gemm_wmma_kernel<<<fblocks, T, 0, stream>>>(
        Up, 0, 128, fw_b + (size_t)si * 128, 384,
        (si == 0) ? fb : (const float*)nullptr,
        Y1, 128, 65536, 128, (si == 0) ? 4 : 3);
  }

  // 6) final 1x1 conv: f32 A operand, f32 transposed store into d_out [C][H*W]
  gemm_wmma_kernel<<<fblocks, T, 0, stream>>>(
      Y1, 1, 128, ow_b, 128, ob, out, 65536, 65536, 128, 2);
}
